// PLM_Graph_40733469836023
// MI455X (gfx1250) — matrix operand order repaired and checked by reading, not compiled, not run on hardware
//
#include <hip/hip_runtime.h>
#include <stdint.h>

// Problem dims (fixed by reference)
#define B_DIM 4096
#define L_DIM 4096
#define H_DIM 1024

// GEMM tiling
#define MT 128      // block tile M
#define NT 128      // block tile N
#define KC 32       // K chunk == bf16 WMMA K
#define LDSPAD 40   // row stride in bf16 elems (80 B: 16B-aligned, bank-skewed)

// ---- TDM availability / arity discrimination --------------------------------
#if defined(__gfx1250__) && __has_builtin(__builtin_amdgcn_tensor_load_to_lds)
#define PLM_HAVE_TDM 1
#else
#define PLM_HAVE_TDM 0
#endif

#if __has_include(<hip/amd_detail/amd_gfx1250_TDM.h>)
#define PLM_TDM_6ARG 1   // therock/clang-23 headers present -> 6-arg builtin
#else
#define PLM_TDM_6ARG 0   // ROCm 7.2 / clang-22 -> 5-arg builtin
#endif

typedef __attribute__((ext_vector_type(16))) __bf16 v16bf;
typedef __attribute__((ext_vector_type(8)))  __bf16 v8bf;
typedef __attribute__((ext_vector_type(4)))  __bf16 v4bf;
typedef __attribute__((ext_vector_type(8)))  float  v8f;
typedef uint32_t u32x4 __attribute__((ext_vector_type(4)));
typedef int      i32x4 __attribute__((ext_vector_type(4)));
typedef int      i32x8 __attribute__((ext_vector_type(8)));

union FragU { v16bf v; v8bf h[2]; };

// A-matrix 16x32 bf16 fragment (ISA 7.12.2): lanes 0-15 hold K 0..7 | 16..23,
// lanes 16-31 hold K 8..15 | 24..31  -> two 16-byte LDS loads per lane.
__device__ inline v16bf lds_frag_a(const __bf16* rowbase, int g) {
  FragU f;
  f.h[0] = *(const v8bf*)(rowbase + g * 8);
  f.h[1] = *(const v8bf*)(rowbase + g * 8 + 16);
  return f.v;
}
// B-matrix 32x16 bf16 fragment: lane group g holds K 8g..8g+7 | 16+8g..16+8g+7.
__device__ inline v16bf lds_frag_b(const __bf16* rowbase, int g) {
  FragU f;
  f.h[0] = *(const v8bf*)(rowbase + g * 8);
  f.h[1] = *(const v8bf*)(rowbase + g * 8 + 16);
  return f.v;
}

// ---------------------------------------------------------------------------
// TDM: one 2D descriptor moves a 128-row x 32-elem bf16 tile (row stride
// H_DIM elems) from global into LDS, inserting 16 B of padding after every
// 64 B row so the LDS image lands directly in the [row][LDSPAD] layout.
//   group1: data_size=1(2B) | pad_enable | pad_interval=3(16 DW) |
//           pad_amount=3(4 DW); tensor_dim0=1024; tensor_dim1=4096;
//           tile_dim0=32; tile_dim1=128; tensor_dim0_stride=1024.
// ---------------------------------------------------------------------------
__device__ inline void tdm_load_tile(uint64_t gaddr, uint32_t ldsaddr) {
#if PLM_HAVE_TDM
  u32x4 g0 = {
      1u,                                                     // count=1 (valid user D#)
      ldsaddr,                                                // lds_addr [63:32]
      (uint32_t)gaddr,                                        // global_addr [95:64]
      (uint32_t)((gaddr >> 32) & 0x01FFFFFFu) | (2u << 30)    // gaddr[56:32] | type=2
  };
  i32x8 g1 = {
      (int)((1u << 16) | (1u << 20) | (3u << 22) | (3u << 25)),  // dsize|pad cfg
      (int)(0x0400u << 16),   // tensor_dim0[15:0]=1024 in bits[63:48]
      (int)(0x1000u << 16),   // tensor_dim1[15:0]=4096 in bits[95:80]
      (int)(32u << 16),       // tile_dim0=32 in bits[127:112]
      (int)128,               // tile_dim1=128 in bits[143:128]
      (int)1024,              // tensor_dim0_stride[31:0]=1024
      0, 0
  };
  i32x4 z4 = {0, 0, 0, 0};
#if PLM_TDM_6ARG
  i32x8 z8 = {0, 0, 0, 0, 0, 0, 0, 0};
  __builtin_amdgcn_tensor_load_to_lds(g0, g1, z4, z4, z8, 0);
#else
  __builtin_amdgcn_tensor_load_to_lds(g0, g1, z4, z4, 0);
#endif
#else
  (void)gaddr; (void)ldsaddr;
#endif
}

__device__ inline void tdm_wait4() {
#if PLM_HAVE_TDM
  __builtin_amdgcn_s_wait_tensorcnt(4);   // retire older chunk's 4 descriptors
#endif
}
__device__ inline void tdm_wait0() {
#if PLM_HAVE_TDM
  __builtin_amdgcn_s_wait_tensorcnt(0);
#endif
}

// ---------------------------------------------------------------------------
// Kernel: dbias[j] = sum_k E[j,k]*W[j,k] + b[j]   (one wave32 per row)
// ---------------------------------------------------------------------------
__global__ void plm_diag_bias_kernel(const float* __restrict__ E,
                                     const float* __restrict__ W,
                                     const float* __restrict__ b,
                                     float* __restrict__ dbias) {
  const int row  = blockIdx.x * 8 + (threadIdx.x >> 5);
  const int lane = threadIdx.x & 31;
  const float4* e4 = (const float4*)(E + (size_t)row * H_DIM);
  const float4* w4 = (const float4*)(W + (size_t)row * H_DIM);
  float s = 0.0f;
#pragma unroll
  for (int i = lane; i < H_DIM / 4; i += 32) {
    float4 e = e4[i], w = w4[i];
    s += e.x * w.x + e.y * w.y + e.z * w.z + e.w * w.w;
  }
#pragma unroll
  for (int off = 16; off > 0; off >>= 1) s += __shfl_down(s, off, 32);
  if (lane == 0) dbias[row] = s + b[row];
}

// ---------------------------------------------------------------------------
// Kernel: one-time f32 -> (bf16 hi, bf16 lo) split (Markidis decomposition)
// ---------------------------------------------------------------------------
__global__ __launch_bounds__(256)
void plm_split_bf16_kernel(const float* __restrict__ src,
                           __bf16* __restrict__ hi, __bf16* __restrict__ lo) {
  const size_t i = (size_t)blockIdx.x * 256 + threadIdx.x;  // float4 index
  float4 v = ((const float4*)src)[i];
  const float f[4] = {v.x, v.y, v.z, v.w};
  v4bf hv, lv;
#pragma unroll
  for (int j = 0; j < 4; ++j) {
    __bf16 h = (__bf16)f[j];
    hv[j] = h;
    lv[j] = (__bf16)(f[j] - (float)h);
  }
  ((v4bf*)hi)[i] = hv;
  ((v4bf*)lo)[i] = lv;
}

// ---------------------------------------------------------------------------
// Main GEMM: pre-split bf16 planes, double-buffered LDS fed by the Tensor
// Data Mover (4 descriptors/chunk issued by wave 0, TENSORcnt-tracked),
// bf16x3 accumulation on v_wmma_f32_16x16x32_bf16.
// ---------------------------------------------------------------------------
__global__ __launch_bounds__(256)
void plm_gemm_bf16_tdm(const __bf16* __restrict__ xh, const __bf16* __restrict__ xl,
                       const __bf16* __restrict__ wh, const __bf16* __restrict__ wl,
                       const float* __restrict__ dbias, float* __restrict__ out) {
  // [buf][plane: hi/lo][row][k]
  __shared__ __bf16 As[2][2][MT][LDSPAD];
  __shared__ __bf16 Bs[2][2][NT][LDSPAD];

  const int tid  = threadIdx.x;
  const int lane = tid & 31;
  const int wave = tid >> 5;
  const int wm   = wave & 1;   // 2 wave-rows -> 64 M each
  const int wn   = wave >> 1;  // 4 wave-cols -> 32 N each
  const int m0   = blockIdx.y * MT;
  const int n0   = blockIdx.x * NT;
  const int nk   = H_DIM / KC; // 32 chunks

#if PLM_HAVE_TDM
  // Uniform byte addresses of each tile plane's first row for this block.
  const uint64_t aByte[2] = {
      (uint64_t)(uintptr_t)xh + (uint64_t)m0 * H_DIM * 2,
      (uint64_t)(uintptr_t)xl + (uint64_t)m0 * H_DIM * 2};
  const uint64_t bByte[2] = {
      (uint64_t)(uintptr_t)wh + (uint64_t)n0 * H_DIM * 2,
      (uint64_t)(uintptr_t)wl + (uint64_t)n0 * H_DIM * 2};
  uint32_t ldsA[2][2], ldsB[2][2];
#pragma unroll
  for (int bfr = 0; bfr < 2; ++bfr)
#pragma unroll
    for (int p = 0; p < 2; ++p) {
      ldsA[bfr][p] = (uint32_t)(uintptr_t)&As[bfr][p][0][0];
      ldsB[bfr][p] = (uint32_t)(uintptr_t)&Bs[bfr][p][0][0];
    }

  if (wave == 0) {  // prime buffer 0 with chunk 0 (4 descriptors)
#pragma unroll
    for (int p = 0; p < 2; ++p) {
      tdm_load_tile(aByte[p], ldsA[0][p]);
      tdm_load_tile(bByte[p], ldsB[0][p]);
    }
  }
#else
  // Fallback staging map: 8 x 16B per thread per chunk, synchronous.
  int pp[4], rr[4], ss[4];
  const __bf16* aSrc[4];
  const __bf16* bSrc[4];
#pragma unroll
  for (int i = 0; i < 4; ++i) {
    const int idx = tid + i * 256;
    pp[i] = idx >> 9;
    rr[i] = (idx >> 2) & 127;
    ss[i] = idx & 3;
    aSrc[i] = (pp[i] ? xl : xh) + (size_t)(m0 + rr[i]) * H_DIM + ss[i] * 8;
    bSrc[i] = (pp[i] ? wl : wh) + (size_t)(n0 + rr[i]) * H_DIM + ss[i] * 8;
  }
#pragma unroll
  for (int i = 0; i < 4; ++i) {
    *(float4*)&As[0][pp[i]][rr[i]][ss[i] * 8] = *(const float4*)aSrc[i];
    *(float4*)&Bs[0][pp[i]][rr[i]][ss[i] * 8] = *(const float4*)bSrc[i];
  }
#endif

  v8f acc[4][2] = {};
  const int fr = lane & 15;
  const int g  = lane >> 4;

  for (int kc = 0; kc < nk; ++kc) {
    const int cur = kc & 1;

#if PLM_HAVE_TDM
    if (wave == 0) {
      if (kc + 1 < nk) {
        const int nxt = cur ^ 1;
        const uint64_t koff = (uint64_t)(kc + 1) * KC * 2;  // bytes
#pragma unroll
        for (int p = 0; p < 2; ++p) {
          tdm_load_tile(aByte[p] + koff, ldsA[nxt][p]);
          tdm_load_tile(bByte[p] + koff, ldsB[nxt][p]);
        }
        tdm_wait4();   // chunk kc landed (in-order per-wave TDM completion)
      } else {
        tdm_wait0();
      }
    }
#else
    if (kc + 1 < nk) {
      const int nxt  = cur ^ 1;
      const int koff = (kc + 1) * KC;
#pragma unroll
      for (int i = 0; i < 4; ++i) {
        *(float4*)&As[nxt][pp[i]][rr[i]][ss[i] * 8] = *(const float4*)(aSrc[i] + koff);
        *(float4*)&Bs[nxt][pp[i]][rr[i]][ss[i] * 8] = *(const float4*)(bSrc[i] + koff);
      }
    }
#endif
    __syncthreads();   // chunk kc visible to all waves

    v16bf bh[2], bl[2];
#pragma unroll
    for (int ni = 0; ni < 2; ++ni) {
      const int r = wn * 32 + ni * 16 + fr;
      bh[ni] = lds_frag_b(&Bs[cur][0][r][0], g);
      bl[ni] = lds_frag_b(&Bs[cur][1][r][0], g);
    }
#pragma unroll
    for (int mi = 0; mi < 4; ++mi) {
      const int r = wm * 64 + mi * 16 + fr;
      v16bf ah = lds_frag_a(&As[cur][0][r][0], g);
      v16bf al = lds_frag_a(&As[cur][1][r][0], g);
#pragma unroll
      for (int ni = 0; ni < 2; ++ni) {
        // bf16x3: hi*hi + hi*lo + lo*hi, f32 accumulate
        acc[mi][ni] = __builtin_amdgcn_wmma_f32_16x16x32_bf16(
            false, ah, false, bh[ni], (short)0, acc[mi][ni], false, false);
        acc[mi][ni] = __builtin_amdgcn_wmma_f32_16x16x32_bf16(
            false, ah, false, bl[ni], (short)0, acc[mi][ni], false, false);
        acc[mi][ni] = __builtin_amdgcn_wmma_f32_16x16x32_bf16(
            false, al, false, bh[ni], (short)0, acc[mi][ni], false, false);
      }
    }
    __syncthreads();   // reads done before this buffer is overwritten
  }

  // epilogue: C/D layout -> lane = N col, VGPR r = M row (+8 for upper lanes)
  const int nCol = lane & 15;
  const int mOff = (lane >> 4) * 8;
#pragma unroll
  for (int ni = 0; ni < 2; ++ni) {
    const int col = n0 + wn * 32 + ni * 16 + nCol;
    const float dbv = dbias[col];
#pragma unroll
    for (int mi = 0; mi < 4; ++mi) {
      const int row = m0 + wm * 64 + mi * 16 + mOff;
#pragma unroll
      for (int r = 0; r < 8; ++r) {
        out[(size_t)(row + r) * L_DIM + col] = acc[mi][ni][r] + dbv;
      }
    }
  }
}

// ---------------------------------------------------------------------------
// Fallback GEMM: fused f32->bf16 hi/lo split in-kernel,
// used only if d_ws is too small for the pre-split planes.
// ---------------------------------------------------------------------------
__global__ __launch_bounds__(256)
void plm_gemm_bias_fused(const float* __restrict__ X, const float* __restrict__ W,
                         const float* __restrict__ dbias, float* __restrict__ out) {
  __shared__ __bf16 As[2][MT][LDSPAD];
  __shared__ __bf16 Bs[2][NT][LDSPAD];

  const int tid  = threadIdx.x;
  const int lane = tid & 31;
  const int wave = tid >> 5;
  const int wm   = wave & 1;
  const int wn   = wave >> 1;
  const int m0   = blockIdx.y * MT;
  const int n0   = blockIdx.x * NT;

  const int lr = tid >> 1;
  const int lc = (tid & 1) * 16;
  const float* Ag = X + (size_t)(m0 + lr) * H_DIM + lc;
  const float* Bg = W + (size_t)(n0 + lr) * H_DIM + lc;

  v8f acc[4][2] = {};
  float4 ar[4], br[4];
#pragma unroll
  for (int i = 0; i < 4; ++i) {
    ar[i] = *(const float4*)(Ag + i * 4);
    br[i] = *(const float4*)(Bg + i * 4);
  }

  const int nk = H_DIM / KC;
  for (int kc = 0; kc < nk; ++kc) {
    __syncthreads();
#pragma unroll
    for (int i = 0; i < 4; ++i) {
      const float fa[4] = {ar[i].x, ar[i].y, ar[i].z, ar[i].w};
      const float fb[4] = {br[i].x, br[i].y, br[i].z, br[i].w};
#pragma unroll
      for (int j = 0; j < 4; ++j) {
        const int c = lc + i * 4 + j;
        __bf16 ha = (__bf16)fa[j];
        As[0][lr][c] = ha;
        As[1][lr][c] = (__bf16)(fa[j] - (float)ha);
        __bf16 hb = (__bf16)fb[j];
        Bs[0][lr][c] = hb;
        Bs[1][lr][c] = (__bf16)(fb[j] - (float)hb);
      }
    }
    __syncthreads();

    if (kc + 1 < nk) {
      const int k0 = (kc + 1) * KC;
#pragma unroll
      for (int i = 0; i < 4; ++i) {
        ar[i] = *(const float4*)(Ag + k0 + i * 4);
        br[i] = *(const float4*)(Bg + k0 + i * 4);
      }
    }

    const int fr = lane & 15;
    const int g  = lane >> 4;
    v16bf bh[2], bl[2];
#pragma unroll
    for (int ni = 0; ni < 2; ++ni) {
      const int r = wn * 32 + ni * 16 + fr;
      bh[ni] = lds_frag_b(&Bs[0][r][0], g);
      bl[ni] = lds_frag_b(&Bs[1][r][0], g);
    }
#pragma unroll
    for (int mi = 0; mi < 4; ++mi) {
      const int r = wm * 64 + mi * 16 + fr;
      v16bf ah = lds_frag_a(&As[0][r][0], g);
      v16bf al = lds_frag_a(&As[1][r][0], g);
#pragma unroll
      for (int ni = 0; ni < 2; ++ni) {
        acc[mi][ni] = __builtin_amdgcn_wmma_f32_16x16x32_bf16(
            false, ah, false, bh[ni], (short)0, acc[mi][ni], false, false);
        acc[mi][ni] = __builtin_amdgcn_wmma_f32_16x16x32_bf16(
            false, ah, false, bl[ni], (short)0, acc[mi][ni], false, false);
        acc[mi][ni] = __builtin_amdgcn_wmma_f32_16x16x32_bf16(
            false, al, false, bh[ni], (short)0, acc[mi][ni], false, false);
      }
    }
  }

  const int nCol = lane & 15;
  const int mOff = (lane >> 4) * 8;
#pragma unroll
  for (int ni = 0; ni < 2; ++ni) {
    const int col = n0 + wn * 32 + ni * 16 + nCol;
    const float dbv = dbias[col];
#pragma unroll
    for (int mi = 0; mi < 4; ++mi) {
      const int row = m0 + wm * 64 + mi * 16 + mOff;
#pragma unroll
      for (int r = 0; r < 8; ++r) {
        out[(size_t)(row + r) * L_DIM + col] = acc[mi][ni][r] + dbv;
      }
    }
  }
}

// ---------------------------------------------------------------------------
extern "C" void kernel_launch(void* const* d_in, const int* in_sizes, int n_in,
                              void* d_out, int out_size, void* d_ws, size_t ws_size,
                              hipStream_t stream) {
  const float* x = (const float*)d_in[0];  // bert_output [B,H]
  const float* E = (const float*)d_in[1];  // label_embed [L,H]
  const float* W = (const float*)d_in[2];  // W [L,H]
  const float* b = (const float*)d_in[3];  // b [L]
  float* out = (float*)d_out;

  const size_t NE         = (size_t)L_DIM * H_DIM;        // 4M elems per matrix
  const size_t biasBytes  = (size_t)L_DIM * sizeof(float);
  const size_t planeBytes = NE * sizeof(__bf16);          // 8 MB per plane
  float* dbias = (float*)d_ws;

  plm_diag_bias_kernel<<<L_DIM / 8, 256, 0, stream>>>(E, W, b, dbias);

  dim3 grid(L_DIM / NT, B_DIM / MT);  // 32 x 32 blocks

  if (ws_size >= biasBytes + 4 * planeBytes) {
    __bf16* xh = (__bf16*)((char*)d_ws + biasBytes);
    __bf16* xl = xh + NE;
    __bf16* wh = xl + NE;
    __bf16* wl = wh + NE;
    const int splitBlocks = (int)(NE / 4 / 256);  // 4096
    plm_split_bf16_kernel<<<splitBlocks, 256, 0, stream>>>(x, xh, xl);
    plm_split_bf16_kernel<<<splitBlocks, 256, 0, stream>>>(W, wh, wl);
    plm_gemm_bf16_tdm<<<grid, 256, 0, stream>>>(xh, xl, wh, wl, dbias, out);
  } else {
    plm_gemm_bias_fused<<<grid, 256, 0, stream>>>(x, W, dbias, out);
  }
}
